// Attention_65094524338925
// MI455X (gfx1250) — matrix operand order repaired and checked by reading, not compiled, not run on hardware
//
#include <hip/hip_runtime.h>
#include <hip/hip_bf16.h>

// ---------------------------------------------------------------------------
// Fused attention scoring for MI455X (gfx1250, wave32, WMMA, 320KB LDS/WGP).
//   B=64, S=2048, H=256.
//   K1: W_text fp32 [k][h] -> bf16 transposed Wt[h][k] (WMMA B-operand order).
//   K2: stage Wt into LDS in a K-interleaved layout (each lane's 8 B-fragments
//       in one 4240B window -> all ds_loads are base+immediate), then two
//       passes (text, aspect) of tanh(X @ W^T) . W_combine -> scores [B*S].
//       Inner loop: 16 ds_load_b128 burst | sched_barrier(0) | one
//       s_wait_dscnt | 8 back-to-back v_wmma_f32_16x16x32_bf16. <256 VGPRs.
//   K3a: softmax over S -> weight [B,1,S]
//   K3b: weighted partial sums over S-chunks (1024 blocks, full HBM parallelism)
//   K3c: deterministic chunk reduction -> out [B,1,H]
// HBM traffic ~ text x2 + aspect x1 ~ 400MB -> ~17us floor at 23.3 TB/s.
// ---------------------------------------------------------------------------

typedef __attribute__((ext_vector_type(16))) __bf16 v16bf;
typedef __attribute__((ext_vector_type(8)))  float  v8f;

#define BDIM 64
#define SDIM 2048
#define HDIM 256
#define NCHUNK 16
#define CHUNK  (SDIM / NCHUNK)          // 128
#define RPITCH 528                      // bytes per 512B row slot (16B pad)
#define LPITCH (8 * RPITCH + 16)        // 4240B per lane group: 1060 dwords,
                                        // == 36 mod 64 banks -> lanes 0-15 on
                                        // distinct banks, 2-way across halves
#define WC_OFF (32 * LPITCH)            // 135680: W_combine LDS offset
#define LDS_BYTES (WC_OFF + 2 * HDIM * 4)

// Branch-free tanh: hardware v_tanh_f32 when clang exposes it, else exp-based.
__device__ __forceinline__ float fast_tanh(float x) {
#if __has_builtin(__builtin_amdgcn_tanhf)
    return __builtin_amdgcn_tanhf(x);
#else
    const float ax = __builtin_fabsf(x);
    const float e  = __expf(-2.0f * ax);
    const float r  = (1.0f - e) / (1.0f + e);
    return __builtin_copysignf(r, x);
#endif
}

// ---- K1: transpose + fp32->bf16 convert of W_text ---------------------------
__global__ void prep_wt_kernel(const float* __restrict__ W, __bf16* __restrict__ Wt) {
    int i = blockIdx.x * blockDim.x + threadIdx.x;
    if (i < HDIM * HDIM) {
        int h = i >> 8;
        int e = i & 255;
        int l = h & 31, kb = h >> 5, s = e & 15, hh = l >> 4;
        int kk = kb * 32 + ((s < 8) ? (8 * hh + s) : (16 + 8 * hh + (s - 8)));
        int j = (e >> 4) * 16 + (l & 15);
        Wt[i] = (__bf16)W[j * HDIM + kk];
    }
}

// 16-bit A-fragment (16x32 bf16). Lanes 0-15: M=lane, K={0..7}U{16..23};
// lanes 16-31: M=lane-16, K={8..15}U{24..31}. p already offset by k*32+half*8.
__device__ __forceinline__ v16bf load_a_frag(const float* __restrict__ p) {
    float4 f0 = *(const float4*)(p);
    float4 f1 = *(const float4*)(p + 4);
    float4 f2 = *(const float4*)(p + 16);
    float4 f3 = *(const float4*)(p + 20);
    v16bf a;
    a[0]  = (__bf16)f0.x; a[1]  = (__bf16)f0.y; a[2]  = (__bf16)f0.z; a[3]  = (__bf16)f0.w;
    a[4]  = (__bf16)f1.x; a[5]  = (__bf16)f1.y; a[6]  = (__bf16)f1.z; a[7]  = (__bf16)f1.w;
    a[8]  = (__bf16)f2.x; a[9]  = (__bf16)f2.y; a[10] = (__bf16)f2.z; a[11] = (__bf16)f2.w;
    a[12] = (__bf16)f3.x; a[13] = (__bf16)f3.y; a[14] = (__bf16)f3.z; a[15] = (__bf16)f3.w;
    return a;
}

// ---- K2: fused GEMM(x2) + tanh + combine-dot -> scores ----------------------
// 256 threads = 8 waves; each wave computes 16 rows' scores. Wt staged in LDS.
__global__ __launch_bounds__(256, 1)
void scores_kernel(const float* __restrict__ text,
                   const float* __restrict__ aspect,
                   const __bf16* __restrict__ Wt,     // [h][k] bf16 global
                   const float* __restrict__ Wc,      // [1, 2H]
                   float* __restrict__ scores) {      // [B*S]
    extern __shared__ __align__(16) char smem[];

    const int tid  = threadIdx.x;
    const int lane = tid & 31;
    const int wave = tid >> 5;
    const int col  = lane & 15;
    const int half = lane >> 4;
    const long rbase = (long)blockIdx.x * 128 + (long)wave * 16;
    const long row   = rbase + col;

    // Stage Wt -> LDS, K-interleaved: global row h (=GEMM K index) lands at
    // lane-group (h%32), slot (h/32). A lane's 8 fragments are then contiguous
    // in one LPITCH window, reachable with immediate DS offsets.
    {
        const uint4* src = (const uint4*)Wt;          // 256 rows x 32 uint4
        for (int idx = tid; idx < HDIM * 32; idx += 256) {
            const int h = idx >> 5;
            const int c = idx & 31;
            *(uint4*)(smem + (size_t)(h & 31) * LPITCH
                           + (size_t)(h >> 5) * RPITCH + (size_t)c * 16) = src[idx];
        }
        float* wclw = (float*)(smem + WC_OFF);
        for (int idx = tid; idx < 2 * HDIM; idx += 256)
            wclw[idx] = Wc[idx];
    }
    __syncthreads();

    const float* wcl   = (const float*)(smem + WC_OFF);
    const char*  lbase = smem + (size_t)lane * LPITCH;   // this lane's fragments

    float partial[8];
#pragma unroll
    for (int j = 0; j < 8; ++j) partial[j] = 0.0f;

    // Two sequential passes (text, then aspect) sharing the LDS-resident Wt.
#pragma unroll 1
    for (int m = 0; m < 2; ++m) {
        const float* src = (m == 0) ? text : aspect;

        // Preload this pass's A fragments: 8 x v16bf = 64 VGPRs.
        const float* rp = src + row * HDIM;
        v16bf A[8];
#pragma unroll
        for (int k = 0; k < 8; ++k)
            A[k] = load_a_frag(rp + k * 32 + half * 8);

#pragma unroll 1
        for (int n = 0; n < 16; ++n) {
            // Burst: 16 ds_load_b128, all base + immediate (k*528 + n*32 + {0,16}).
            v16bf bf[8];
#pragma unroll
            for (int k = 0; k < 8; ++k)
                bf[k] = *(const v16bf*)(lbase + k * RPITCH + n * 32);
#if __has_builtin(__builtin_amdgcn_sched_barrier)
            // Hard fence (mask 0 = nothing crosses): all 8 B fragments must be
            // live here -> distinct registers, single s_wait_dscnt, and the
            // WMMA chain below runs with no interleaved ds-writes (no WAR NOPs).
            __builtin_amdgcn_sched_barrier(0);
#endif
            v8f acc0 = {};
            v8f acc1 = {};
#pragma unroll
            for (int k = 0; k < 8; k += 2) {
                acc0 = __builtin_amdgcn_wmma_f32_16x16x32_bf16(
                           false, A[k],     false, bf[k],     (short)0, acc0, false, false);
                acc1 = __builtin_amdgcn_wmma_f32_16x16x32_bf16(
                           false, A[k + 1], false, bf[k + 1], (short)0, acc1, false, false);
            }
            // C/D layout: VGPR j holds row (j + 8*half), column col (+n*16).
            const float wc = wcl[m * HDIM + n * 16 + col];
#pragma unroll
            for (int j = 0; j < 8; ++j)
                partial[j] += fast_tanh(acc0[j] + acc1[j]) * wc;
        }
    }

    // Reduce the 16 columns (lanes 0-15 own rows 0-7, lanes 16-31 rows 8-15).
#pragma unroll
    for (int j = 0; j < 8; ++j) {
        float v = partial[j];
        v += __shfl_xor(v, 1);
        v += __shfl_xor(v, 2);
        v += __shfl_xor(v, 4);
        v += __shfl_xor(v, 8);
        partial[j] = v;
    }
    if (col == 0) {
#pragma unroll
        for (int j = 0; j < 8; ++j)
            scores[rbase + half * 8 + j] = partial[j];
    }
}

// ---- K3a: softmax over S -> weight ------------------------------------------
__global__ __launch_bounds__(256)
void softmax_kernel(const float* __restrict__ scores,
                    float* __restrict__ out_w) {      // [B*S]
    __shared__ float red[256];
    __shared__ float wsh[SDIM];
    const int b   = blockIdx.x;
    const int tid = threadIdx.x;
    const float* sc = scores + (size_t)b * SDIM;

    float m = -3.402823466e38f;
    for (int s = tid; s < SDIM; s += 256) m = fmaxf(m, sc[s]);
    red[tid] = m; __syncthreads();
    for (int off = 128; off > 0; off >>= 1) {
        if (tid < off) red[tid] = fmaxf(red[tid], red[tid + off]);
        __syncthreads();
    }
    const float mx = red[0];
    __syncthreads();

    float sum = 0.0f;
    for (int s = tid; s < SDIM; s += 256) {
        float e = __expf(sc[s] - mx);
        wsh[s] = e;
        sum += e;
    }
    red[tid] = sum; __syncthreads();
    for (int off = 128; off > 0; off >>= 1) {
        if (tid < off) red[tid] += red[tid + off];
        __syncthreads();
    }
    const float inv = 1.0f / red[0];
    __syncthreads();

    for (int s = tid; s < SDIM; s += 256)
        out_w[(size_t)b * SDIM + s] = wsh[s] * inv;
}

// ---- K3b: partial weighted sums over S-chunks -------------------------------
// grid = B*NCHUNK blocks; block (b,c) reduces CHUNK rows of text.
__global__ __launch_bounds__(256)
void partial_out_kernel(const float* __restrict__ out_w,
                        const float* __restrict__ text,
                        float* __restrict__ partial) { // [B*NCHUNK*H]
    __shared__ float wsh[CHUNK];
    const int b   = blockIdx.x >> 4;
    const int c   = blockIdx.x & (NCHUNK - 1);
    const int tid = threadIdx.x;
    const int s0  = c * CHUNK;

    if (tid < CHUNK) wsh[tid] = out_w[(size_t)b * SDIM + s0 + tid];
    __syncthreads();

    const float* tb = text + ((size_t)b * SDIM + s0) * HDIM;
    float acc = 0.0f;
#pragma unroll 4
    for (int s = 0; s < CHUNK; ++s)
        acc += wsh[s] * tb[(size_t)s * HDIM + tid];
    partial[((size_t)b * NCHUNK + c) * HDIM + tid] = acc;
}

// ---- K3c: deterministic reduction of chunk partials -> out ------------------
__global__ __launch_bounds__(256)
void reduce_out_kernel(const float* __restrict__ partial,
                       float* __restrict__ out_v) {   // [B*H]
    const int b   = blockIdx.x;
    const int tid = threadIdx.x;
    float acc = 0.0f;
#pragma unroll
    for (int c = 0; c < NCHUNK; ++c)
        acc += partial[((size_t)b * NCHUNK + c) * HDIM + tid];
    out_v[(size_t)b * HDIM + tid] = acc;
}

// ---------------------------------------------------------------------------
extern "C" void kernel_launch(void* const* d_in, const int* in_sizes, int n_in,
                              void* d_out, int out_size, void* d_ws, size_t ws_size,
                              hipStream_t stream) {
    const float* text   = (const float*)d_in[0];   // [B,S,H] fp32
    const float* aspect = (const float*)d_in[1];   // [B,S,H] fp32
    const float* Wtext  = (const float*)d_in[2];   // [H,H]   fp32
    const float* Wc     = (const float*)d_in[3];   // [1,2H]  fp32

    // ws layout: [Wt bf16 128KB][scores fp32 512KB][partial fp32 1MB]
    __bf16* Wt     = (__bf16*)d_ws;
    float*  scores = (float*)((char*)d_ws + (size_t)HDIM * HDIM * sizeof(__bf16));
    float*  part   = scores + (size_t)BDIM * SDIM;

    float* out_w = (float*)d_out;                    // weight [B,1,S] flat
    float* out_v = out_w + (size_t)BDIM * SDIM;      // out    [B,1,H] flat

    prep_wt_kernel<<<(HDIM * HDIM + 255) / 256, 256, 0, stream>>>(Wtext, Wt);
    scores_kernel<<<(BDIM * SDIM) / 128, 256, LDS_BYTES, stream>>>(
        text, aspect, Wt, Wc, scores);
    softmax_kernel<<<BDIM, 256, 0, stream>>>(scores, out_w);
    partial_out_kernel<<<BDIM * NCHUNK, 256, 0, stream>>>(out_w, text, part);
    reduce_out_kernel<<<BDIM, 256, 0, stream>>>(part, out_v);
}
